// SGLC_Classifier_41455024341474
// MI455X (gfx1250) — compile-verified, matrix operand order checked by reading, not run on hardware
//
#include <hip/hip_runtime.h>
#include <hip/hip_bf16.h>
#include <math.h>

#define TT 16
#define BB 32
#define NNODE 256
#define DD 128
#define HH 16
#define GSTEPS 5
#define CLS 2
#define SKIPC 0.3f
#define EPSC 1e-8f

typedef __bf16 bf16_t;
typedef __attribute__((ext_vector_type(16))) __bf16 v16bf;
typedef __attribute__((ext_vector_type(8)))  float  v8f;

// ---------------------------------------------------------------------------
// WMMA fragment helpers (CDNA5 16x16x32 bf16 layouts, ISA 7.12.2)
// ---------------------------------------------------------------------------
__device__ __forceinline__ int frag_k(int lane, int e) {
    // element e of the per-lane v16bf -> K index within the 32-wide K slab
    return (e & 7) | ((lane >> 4) << 3) | ((e & 8) << 1);
}

// A-style fragment: 16 rows (M = lane&15) x 32 K, from row-major src (ld = row stride).
// Also used for B^T sources stored row-major as [N][K]: contiguous -> global_load_b128.
__device__ __forceinline__ v16bf load_frag_rm(const bf16_t* __restrict__ src, int ld) {
    int lane = threadIdx.x & 31;
    int row  = lane & 15;
    v16bf f;
#pragma unroll
    for (int e = 0; e < 16; ++e)
        f[e] = src[row * ld + frag_k(lane, e)];
    return f;
}

#define WMMA_BF16(a, b, c) \
    __builtin_amdgcn_wmma_f32_16x16x32_bf16(false, (a), false, (b), (short)0, (c), false, false)

__device__ __forceinline__ float wave_sum(float v) {
#pragma unroll
    for (int m = 16; m >= 1; m >>= 1) v += __shfl_xor(v, m, 32);
    return v;
}

__device__ __forceinline__ float sigmoidf(float x) { return 1.0f / (1.0f + __expf(-x)); }

// C/D tile store helper: row = 8*(lane>>4)+v, col = lane&15
__device__ __forceinline__ void store_tile_f32(float* __restrict__ o, int ld, v8f acc,
                                               float scale) {
    int lane = threadIdx.x & 31;
    int cn = lane & 15, mb = (lane >> 4) * 8;
#pragma unroll
    for (int v = 0; v < 8; ++v)
        o[(mb + v) * ld + cn] = acc[v] * scale;
}

__device__ __forceinline__ void store_tile_bf16(bf16_t* __restrict__ o, int ld, v8f acc,
                                                float bias) {
    int lane = threadIdx.x & 31;
    int cn = lane & 15, mb = (lane >> 4) * 8;
#pragma unroll
    for (int v = 0; v < 8; ++v)
        o[(mb + v) * ld + cn] = (bf16_t)(acc[v] + bias);
}

// ---------------------------------------------------------------------------
// Weight transpose fp32 [K x N] -> bf16 [N x K]
// ---------------------------------------------------------------------------
__global__ void k_transpose_bf16(const float* __restrict__ W, bf16_t* __restrict__ Wt,
                                 int K, int N) {
    int idx = blockIdx.x * blockDim.x + threadIdx.x;
    if (idx >= K * N) return;
    int k = idx / N, n = idx % N;
    Wt[n * K + k] = (bf16_t)W[k * N + n];
}

// ---------------------------------------------------------------------------
// Multi-head weighted-cosine rows: xwn[h,b,n,:] = normalize(x[b,n,:]*Wgl[h,:])  (bf16)
// one wave per (h,b,n)
// ---------------------------------------------------------------------------
__global__ void k_xw_norm(const float* __restrict__ x, const float* __restrict__ Wgl,
                          bf16_t* __restrict__ xwn) {
    int wave = blockIdx.x * (blockDim.x >> 5) + (threadIdx.x >> 5);
    int lane = threadIdx.x & 31;
    int n  = wave % NNODE;
    int hb = wave / NNODE;
    int b  = hb % BB;
    int h  = hb / BB;
    if (h >= HH) return;
    const float* xr = x + ((size_t)b * NNODE + n) * DD;
    const float* wg = Wgl + h * DD;
    float vals[4];
    float ss = 0.0f;
#pragma unroll
    for (int i = 0; i < 4; ++i) {
        int d = lane + 32 * i;
        float v = xr[d] * wg[d];
        vals[i] = v;
        ss += v * v;
    }
    ss = wave_sum(ss);
    float rinv = 1.0f / (sqrtf(ss) + EPSC);
    bf16_t* o = xwn + (((size_t)h * BB + b) * NNODE + n) * DD;
#pragma unroll
    for (int i = 0; i < 4; ++i) o[lane + 32 * i] = (bf16_t)(vals[i] * rinv);
}

// ---------------------------------------------------------------------------
// att tile (16 x 32 macro-tile/wave): (1/H) sum_h xwn[h,b] @ xwn[h,b]^T
// A fragment shared by both N-tiles. 128 WMMAs / wave.
// ---------------------------------------------------------------------------
__global__ void __launch_bounds__(32) k_attn(const bf16_t* __restrict__ xwn,
                                             float* __restrict__ att) {
    int i = blockIdx.x, j = blockIdx.y, b = blockIdx.z;   // j over NNODE/32
    v8f acc0 = {}, acc1 = {};
    for (int h = 0; h < HH; ++h) {
        const bf16_t* base = xwn + (((size_t)h * BB + b) * NNODE) * DD;
        const bf16_t* arow = base + (size_t)i * 16 * DD;
        const bf16_t* brow0 = base + (size_t)(j * 32) * DD;
        const bf16_t* brow1 = base + (size_t)(j * 32 + 16) * DD;
#pragma unroll
        for (int kb = 0; kb < DD / 32; ++kb) {
            v16bf fa  = load_frag_rm(arow + kb * 32, DD);
            v16bf fb0 = load_frag_rm(brow0 + kb * 32, DD);   // B^T (row-major NxK)
            v16bf fb1 = load_frag_rm(brow1 + kb * 32, DD);
            acc0 = WMMA_BF16(fa, fb0, acc0);
            acc1 = WMMA_BF16(fa, fb1, acc1);
        }
    }
    const float scale = 1.0f / (float)HH;
    float* o = att + ((size_t)b * NNODE + i * 16) * NNODE + j * 32;
    store_tile_f32(o, NNODE, acc0, scale);
    store_tile_f32(o + 16, NNODE, acc1, scale);
}

// ---------------------------------------------------------------------------
// A_new = SKIP*A + (1-SKIP) * relu(att) / rowsum ; writes f32 + bf16. wave/row.
// ---------------------------------------------------------------------------
__global__ void k_att_norm(const float* __restrict__ att, float* __restrict__ A,
                           bf16_t* __restrict__ Abf) {
    int wave = blockIdx.x * (blockDim.x >> 5) + (threadIdx.x >> 5);
    int lane = threadIdx.x & 31;
    int n = wave % NNODE, b = wave / NNODE;
    if (b >= BB) return;
    const float* row  = att + ((size_t)b * NNODE + n) * NNODE;
    float*       arow = A   + ((size_t)b * NNODE + n) * NNODE;
    bf16_t*      brow = Abf + ((size_t)b * NNODE + n) * NNODE;
    float vals[8];
    float s = 0.0f;
#pragma unroll
    for (int i = 0; i < 8; ++i) {
        float v = row[lane + 32 * i];
        v = v > 0.0f ? v : 0.0f;
        vals[i] = v;
        s += v;
    }
    s = wave_sum(s);
    float inv = (1.0f - SKIPC) / (s + EPSC);
#pragma unroll
    for (int i = 0; i < 8; ++i) {
        int m = lane + 32 * i;
        float an = SKIPC * arow[m] + vals[i] * inv;
        arow[m] = an;
        brow[m] = (bf16_t)an;
    }
}

// ---------------------------------------------------------------------------
// h init per timestep: h_f32 = x_t, h_bf16 = bf16(x_t), h_bf16^T = bf16(x_t)^T
// ---------------------------------------------------------------------------
__global__ void k_init_h(const float* __restrict__ xt, float* __restrict__ hf,
                         bf16_t* __restrict__ hbf, bf16_t* __restrict__ hbfT,
                         int total) {
    int idx = blockIdx.x * blockDim.x + threadIdx.x;
    if (idx >= total) return;
    float v = xt[idx];
    hf[idx] = v;
    hbf[idx] = (bf16_t)v;
    int d = idx % DD;
    int n = (idx / DD) % NNODE;
    int b = idx / (DD * NNODE);
    hbfT[((size_t)b * DD + d) * NNODE + n] = (bf16_t)v;
}

// ---------------------------------------------------------------------------
// tmp = A_new @ h   (K=256). B taken from transposed h (hbfT, [d][node]) so every
// fragment load is contiguous (-> global_load_b128). 16x32 macro-tile / wave.
// ---------------------------------------------------------------------------
__global__ void __launch_bounds__(32) k_gemm_ah(const bf16_t* __restrict__ Abf,
                                               const bf16_t* __restrict__ hbfT,
                                               bf16_t* __restrict__ tmp) {
    int i = blockIdx.x, j = blockIdx.y, b = blockIdx.z;   // j over DD/32
    const bf16_t* Ab = Abf + (size_t)b * NNODE * NNODE + (size_t)i * 16 * NNODE;
    const bf16_t* Ht = hbfT + (size_t)b * DD * NNODE;
    const bf16_t* Bt0 = Ht + (size_t)(j * 32) * NNODE;
    const bf16_t* Bt1 = Ht + (size_t)(j * 32 + 16) * NNODE;
    v8f acc0 = {}, acc1 = {};
#pragma unroll 2
    for (int kb = 0; kb < NNODE / 32; ++kb) {
        __builtin_prefetch(Ab + (kb + 1) * 32, 0, 1);
        v16bf fa  = load_frag_rm(Ab + kb * 32, NNODE);
        v16bf fb0 = load_frag_rm(Bt0 + kb * 32, NNODE);
        v16bf fb1 = load_frag_rm(Bt1 + kb * 32, NNODE);
        acc0 = WMMA_BF16(fa, fb0, acc0);
        acc1 = WMMA_BF16(fa, fb1, acc1);
    }
    bf16_t* o = tmp + (size_t)b * NNODE * DD + ((size_t)i * 16) * DD + j * 32;
    store_tile_bf16(o, DD, acc0, 0.0f);
    store_tile_bf16(o + 16, DD, acc1, 0.0f);
}

// ---------------------------------------------------------------------------
// a = tmp @ W_a + b_a   (K=128), Wat is bf16 [N][K]. 16x32 macro-tile / wave.
// ---------------------------------------------------------------------------
__global__ void __launch_bounds__(32) k_gemm_wa(const bf16_t* __restrict__ tmp,
                                               const bf16_t* __restrict__ Wat,
                                               const float* __restrict__ ba,
                                               bf16_t* __restrict__ abf) {
    int i = blockIdx.x, j = blockIdx.y, b = blockIdx.z;   // j over DD/32
    const bf16_t* src = tmp + (size_t)b * NNODE * DD + (size_t)i * 16 * DD;
    const bf16_t* W0 = Wat + (size_t)(j * 32) * DD;
    const bf16_t* W1 = Wat + (size_t)(j * 32 + 16) * DD;
    v8f acc0 = {}, acc1 = {};
#pragma unroll
    for (int kb = 0; kb < DD / 32; ++kb) {
        v16bf fa  = load_frag_rm(src + kb * 32, DD);
        v16bf fb0 = load_frag_rm(W0 + kb * 32, DD);
        v16bf fb1 = load_frag_rm(W1 + kb * 32, DD);
        acc0 = WMMA_BF16(fa, fb0, acc0);
        acc1 = WMMA_BF16(fa, fb1, acc1);
    }
    int lane = threadIdx.x;
    int cn = lane & 15;
    bf16_t* o = abf + (size_t)b * NNODE * DD + ((size_t)i * 16) * DD + j * 32;
    store_tile_bf16(o, DD, acc0, ba[j * 32 + cn]);
    store_tile_bf16(o + 16, DD, acc1, ba[j * 32 + 16 + cn]);
}

// ---------------------------------------------------------------------------
// gates: r = sig([a,h]@W_r + b_r), z = sig([a,h]@W_z + b_z)
// Wrt/Wzt are bf16 [D][2D]. 16x32 macro-tile, a/h fragments shared across both
// gates and both N-tiles (32 WMMAs / wave).
// outputs: rh_bf = bf16(r*h), z_f32
// ---------------------------------------------------------------------------
__global__ void __launch_bounds__(32) k_gates(const bf16_t* __restrict__ abf,
                                              const bf16_t* __restrict__ hbf,
                                              const float* __restrict__ hf,
                                              const bf16_t* __restrict__ Wrt,
                                              const bf16_t* __restrict__ Wzt,
                                              const float* __restrict__ br,
                                              const float* __restrict__ bz,
                                              bf16_t* __restrict__ rhbf,
                                              float* __restrict__ zf) {
    int i = blockIdx.x, j = blockIdx.y, b = blockIdx.z;   // j over DD/32
    const bf16_t* As = abf + (size_t)b * NNODE * DD + (size_t)i * 16 * DD;
    const bf16_t* Hs = hbf + (size_t)b * NNODE * DD + (size_t)i * 16 * DD;
    const int ld2 = 2 * DD;
    const bf16_t* Wr0 = Wrt + (size_t)(j * 32) * ld2;
    const bf16_t* Wr1 = Wrt + (size_t)(j * 32 + 16) * ld2;
    const bf16_t* Wz0 = Wzt + (size_t)(j * 32) * ld2;
    const bf16_t* Wz1 = Wzt + (size_t)(j * 32 + 16) * ld2;
    v8f ar0 = {}, ar1 = {}, az0 = {}, az1 = {};
#pragma unroll
    for (int kb = 0; kb < DD / 32; ++kb) {
        v16bf fa = load_frag_rm(As + kb * 32, DD);
        v16bf fh = load_frag_rm(Hs + kb * 32, DD);
        ar0 = WMMA_BF16(fa, load_frag_rm(Wr0 + kb * 32, ld2), ar0);
        ar0 = WMMA_BF16(fh, load_frag_rm(Wr0 + DD + kb * 32, ld2), ar0);
        ar1 = WMMA_BF16(fa, load_frag_rm(Wr1 + kb * 32, ld2), ar1);
        ar1 = WMMA_BF16(fh, load_frag_rm(Wr1 + DD + kb * 32, ld2), ar1);
        az0 = WMMA_BF16(fa, load_frag_rm(Wz0 + kb * 32, ld2), az0);
        az0 = WMMA_BF16(fh, load_frag_rm(Wz0 + DD + kb * 32, ld2), az0);
        az1 = WMMA_BF16(fa, load_frag_rm(Wz1 + kb * 32, ld2), az1);
        az1 = WMMA_BF16(fh, load_frag_rm(Wz1 + DD + kb * 32, ld2), az1);
    }
    int lane = threadIdx.x;
    int cn = lane & 15, mb = (lane >> 4) * 8;
#pragma unroll
    for (int tile = 0; tile < 2; ++tile) {
        int ncol = j * 32 + tile * 16 + cn;
        float brv = br[ncol], bzv = bz[ncol];
        v8f ar = tile ? ar1 : ar0;
        v8f az = tile ? az1 : az0;
#pragma unroll
        for (int v = 0; v < 8; ++v) {
            int row = i * 16 + mb + v;
            size_t idx = ((size_t)b * NNODE + row) * DD + ncol;
            float r = sigmoidf(ar[v] + brv);
            float z = sigmoidf(az[v] + bzv);
            float hv = hf[idx];
            rhbf[idx] = (bf16_t)(r * hv);
            zf[idx] = z;
        }
    }
}

// ---------------------------------------------------------------------------
// h_tilde = tanh([a, r*h]@W_h + b_h); h = (1-z)*h + z*h_tilde
// 16x32 macro-tile (16 WMMAs / wave). Writes h (f32), h (bf16) and h^T (bf16).
// ---------------------------------------------------------------------------
__global__ void __launch_bounds__(32) k_htilde(const bf16_t* __restrict__ abf,
                                               const bf16_t* __restrict__ rhbf,
                                               const bf16_t* __restrict__ Wht,
                                               const float* __restrict__ bh,
                                               const float* __restrict__ zf,
                                               float* __restrict__ hf,
                                               bf16_t* __restrict__ hbf,
                                               bf16_t* __restrict__ hbfT) {
    int i = blockIdx.x, j = blockIdx.y, b = blockIdx.z;   // j over DD/32
    const bf16_t* As = abf  + (size_t)b * NNODE * DD + (size_t)i * 16 * DD;
    const bf16_t* Rs = rhbf + (size_t)b * NNODE * DD + (size_t)i * 16 * DD;
    const int ld2 = 2 * DD;
    const bf16_t* W0 = Wht + (size_t)(j * 32) * ld2;
    const bf16_t* W1 = Wht + (size_t)(j * 32 + 16) * ld2;
    v8f acc0 = {}, acc1 = {};
#pragma unroll
    for (int kb = 0; kb < DD / 32; ++kb) {
        v16bf fa = load_frag_rm(As + kb * 32, DD);
        v16bf fr = load_frag_rm(Rs + kb * 32, DD);
        acc0 = WMMA_BF16(fa, load_frag_rm(W0 + kb * 32, ld2), acc0);
        acc0 = WMMA_BF16(fr, load_frag_rm(W0 + DD + kb * 32, ld2), acc0);
        acc1 = WMMA_BF16(fa, load_frag_rm(W1 + kb * 32, ld2), acc1);
        acc1 = WMMA_BF16(fr, load_frag_rm(W1 + DD + kb * 32, ld2), acc1);
    }
    int lane = threadIdx.x;
    int cn = lane & 15, mb = (lane >> 4) * 8;
#pragma unroll
    for (int tile = 0; tile < 2; ++tile) {
        int ncol = j * 32 + tile * 16 + cn;
        float bhv = bh[ncol];
        v8f acc = tile ? acc1 : acc0;
#pragma unroll
        for (int v = 0; v < 8; ++v) {
            int row = i * 16 + mb + v;
            size_t idx = ((size_t)b * NNODE + row) * DD + ncol;
            float ht = tanhf(acc[v] + bhv);
            float z = zf[idx];
            float hn = (1.0f - z) * hf[idx] + z * ht;
            hf[idx] = hn;
            hbf[idx] = (bf16_t)hn;
            hbfT[((size_t)b * DD + ncol) * NNODE + row] = (bf16_t)hn;
        }
    }
}

// ---------------------------------------------------------------------------
// logits[b,c] = h[b].flatten() @ W_fc[:,c] + b_fc[c]   (tiny: plain reduction)
// ---------------------------------------------------------------------------
__global__ void k_fc(const float* __restrict__ hf, const float* __restrict__ Wfc,
                     const float* __restrict__ bfc, float* __restrict__ out) {
    int b = blockIdx.x, c = blockIdx.y;
    __shared__ float red[256];
    const float* hb = hf + (size_t)b * NNODE * DD;
    float s = 0.0f;
    for (int k = threadIdx.x; k < NNODE * DD; k += 256)
        s += hb[k] * Wfc[(size_t)k * CLS + c];
    red[threadIdx.x] = s;
    __syncthreads();
    for (int off = 128; off > 0; off >>= 1) {
        if ((int)threadIdx.x < off) red[threadIdx.x] += red[threadIdx.x + off];
        __syncthreads();
    }
    if (threadIdx.x == 0) out[b * CLS + c] = red[0] + bfc[c];
}

// ---------------------------------------------------------------------------
extern "C" void kernel_launch(void* const* d_in, const int* in_sizes, int n_in,
                              void* d_out, int out_size, void* d_ws, size_t ws_size,
                              hipStream_t stream) {
    const float* x        = (const float*)d_in[0];   // (T,B,N,D)
    const float* supports = (const float*)d_in[1];   // (B,N,N)
    const float* Wgl      = (const float*)d_in[2];   // (H,D)
    const float* Wa       = (const float*)d_in[3];   // (D,D)
    const float* ba       = (const float*)d_in[4];
    const float* Wr       = (const float*)d_in[5];   // (2D,D)
    const float* br       = (const float*)d_in[6];
    const float* Wz       = (const float*)d_in[7];
    const float* bz       = (const float*)d_in[8];
    const float* Wh       = (const float*)d_in[9];
    const float* bh       = (const float*)d_in[10];
    const float* Wfc      = (const float*)d_in[11];  // (N*D, 2)
    const float* bfc      = (const float*)d_in[12];
    float* out            = (float*)d_out;

    char* ws = (char*)d_ws;
    size_t off = 0;
    auto carve = [&](size_t bytes) -> void* {
        void* p = ws + off;
        off = (off + bytes + 255) & ~(size_t)255;
        return p;
    };
    float*  A_f    = (float*) carve((size_t)BB * NNODE * NNODE * 4);
    float*  att_f  = (float*) carve((size_t)BB * NNODE * NNODE * 4);
    bf16_t* A_bf   = (bf16_t*)carve((size_t)BB * NNODE * NNODE * 2);
    bf16_t* xwn    = (bf16_t*)carve((size_t)HH * BB * NNODE * DD * 2);
    float*  h_f    = (float*) carve((size_t)BB * NNODE * DD * 4);
    bf16_t* h_bf   = (bf16_t*)carve((size_t)BB * NNODE * DD * 2);
    bf16_t* h_bfT  = (bf16_t*)carve((size_t)BB * NNODE * DD * 2);
    bf16_t* tmp_bf = (bf16_t*)carve((size_t)BB * NNODE * DD * 2);
    bf16_t* a_bf   = (bf16_t*)carve((size_t)BB * NNODE * DD * 2);
    bf16_t* rh_bf  = (bf16_t*)carve((size_t)BB * NNODE * DD * 2);
    float*  z_f    = (float*) carve((size_t)BB * NNODE * DD * 4);
    bf16_t* Wa_t   = (bf16_t*)carve((size_t)DD * DD * 2);
    bf16_t* Wr_t   = (bf16_t*)carve((size_t)2 * DD * DD * 2);
    bf16_t* Wz_t   = (bf16_t*)carve((size_t)2 * DD * DD * 2);
    bf16_t* Wh_t   = (bf16_t*)carve((size_t)2 * DD * DD * 2);
    (void)ws_size; (void)in_sizes; (void)n_in; (void)out_size;

    // A <- supports
    hipMemcpyAsync(A_f, supports, (size_t)BB * NNODE * NNODE * sizeof(float),
                   hipMemcpyDeviceToDevice, stream);

    // bf16-transpose the GEMM weights: [K x N] -> [N x K]
    k_transpose_bf16<<<(DD * DD + 255) / 256, 256, 0, stream>>>(Wa, Wa_t, DD, DD);
    k_transpose_bf16<<<(2 * DD * DD + 255) / 256, 256, 0, stream>>>(Wr, Wr_t, 2 * DD, DD);
    k_transpose_bf16<<<(2 * DD * DD + 255) / 256, 256, 0, stream>>>(Wz, Wz_t, 2 * DD, DD);
    k_transpose_bf16<<<(2 * DD * DD + 255) / 256, 256, 0, stream>>>(Wh, Wh_t, 2 * DD, DD);

    const int hTotal = BB * NNODE * DD;
    dim3 tile_nd(NNODE / 16, DD / 32, BB);     // (16, 4, 32) -> 16x32 macro-tiles
    dim3 tile_nn(NNODE / 16, NNODE / 32, BB);  // (16, 8, 32) -> 16x32 macro-tiles

    for (int t = 0; t < TT; ++t) {
        const float* xt = x + (size_t)t * BB * NNODE * DD;

        // graph learner
        k_xw_norm<<<HH * BB * NNODE / 8, 256, 0, stream>>>(xt, Wgl, xwn);
        k_attn<<<tile_nn, 32, 0, stream>>>(xwn, att_f);
        k_att_norm<<<BB * NNODE / 8, 256, 0, stream>>>(att_f, A_f, A_bf);

        // GGNN propagation
        k_init_h<<<(hTotal + 255) / 256, 256, 0, stream>>>(xt, h_f, h_bf, h_bfT, hTotal);
        for (int s = 0; s < GSTEPS; ++s) {
            k_gemm_ah<<<tile_nd, 32, 0, stream>>>(A_bf, h_bfT, tmp_bf);
            k_gemm_wa<<<tile_nd, 32, 0, stream>>>(tmp_bf, Wa_t, ba, a_bf);
            k_gates<<<tile_nd, 32, 0, stream>>>(a_bf, h_bf, h_f, Wr_t, Wz_t, br, bz,
                                                rh_bf, z_f);
            k_htilde<<<tile_nd, 32, 0, stream>>>(a_bf, rh_bf, Wh_t, bh, z_f, h_f,
                                                 h_bf, h_bfT);
        }
    }

    // classifier on last timestep's h
    k_fc<<<dim3(BB, CLS), 256, 0, stream>>>(h_f, Wfc, bfc, out);
}
